// KMeans_8727373546246
// MI455X (gfx1250) — matrix operand order repaired and checked by reading, not compiled, not run on hardware
//
#include <hip/hip_runtime.h>

typedef __bf16 bf16;
typedef __attribute__((ext_vector_type(16))) __bf16 v16bf;
typedef __attribute__((ext_vector_type(8)))  float  v8f;

#define NPTS   262144
#define KCENT  1024
#define DIM    64
#define CHUNK  128          // centers staged per LDS chunk
#define LDPAD  72           // padded bf16 row stride (144B = 36 banks -> conflict-free)
#define ROWS_PER_BLOCK 256  // 8 waves * 32 points (2 M-tiles per wave)

__global__ __launch_bounds__(256) void kmeans_assign_kernel(
    const float* __restrict__ x,
    const float* __restrict__ centers,
    int* __restrict__ out) {

  __shared__ bf16  s_hi[CHUNK][LDPAD];
  __shared__ bf16  s_lo[CHUNK][LDPAD];
  __shared__ float s_pn[CHUNK * 2];   // per-half center-norm partials

  const int tid    = threadIdx.x;
  const int wave   = tid >> 5;
  const int lane   = tid & 31;
  const int l16    = lane & 15;
  const bool upper = (lane >= 16);
  const int rbase  = upper ? 8 : 0;

  // ---------------- Load this wave's 2 M-tiles (32 points) as bf16 hi/lo ----
  // A 16x32 bf16 layout: M = lane%16; lanes 0-15 hold K k0+0..15,
  // lanes 16-31 hold K k0+16..31.
  const int rowbase = blockIdx.x * ROWS_PER_BLOCK + wave * 32;
  const int koff    = upper ? 16 : 0;

  v16bf a_hi[2][2], a_lo[2][2];   // [m-tile][k-chunk]
  float xn[2][8];
#pragma unroll
  for (int mt = 0; mt < 2; ++mt) {
    const float* xr = x + (size_t)(rowbase + mt * 16 + l16) * DIM;
    float partial = 0.f;
#pragma unroll
    for (int kc = 0; kc < 2; ++kc) {
      const float4* p = (const float4*)(xr + kc * 32 + koff);        // K = k0+koff..+7
      const float4* q = (const float4*)(xr + kc * 32 + 8 + koff);    // K = k0+8+koff..+7
      float v[16];
      float4 t0 = p[0], t1 = p[1], t2 = q[0], t3 = q[1];
      v[0]=t0.x; v[1]=t0.y; v[2]=t0.z; v[3]=t0.w;
      v[4]=t1.x; v[5]=t1.y; v[6]=t1.z; v[7]=t1.w;
      v[8]=t2.x; v[9]=t2.y; v[10]=t2.z; v[11]=t2.w;
      v[12]=t3.x; v[13]=t3.y; v[14]=t3.z; v[15]=t3.w;
#pragma unroll
      for (int i = 0; i < 16; ++i) {
        bf16 h = (bf16)v[i];
        bf16 l = (bf16)(v[i] - (float)h);
        a_hi[mt][kc][i] = h;
        a_lo[mt][kc][i] = l;
        partial += v[i] * v[i];
      }
    }
    // Full ||x||^2 per row: lane L and L+16 each hold half the K values.
    float xn_full = partial + __shfl_xor(partial, 16, 32);
#pragma unroll
    for (int r = 0; r < 8; ++r) xn[mt][r] = __shfl(xn_full, rbase + r, 32);
  }

  float best[2][8];
  int   bidx[2][8];
#pragma unroll
  for (int mt = 0; mt < 2; ++mt)
#pragma unroll
    for (int r = 0; r < 8; ++r) { best[mt][r] = 3.4e38f; bidx[mt][r] = 0; }

  const int bkoff = upper ? 16 : 0;  // B 32x16: lanes 0-15 K=0..15, lanes 16-31 K=16..31

  for (int chunk = 0; chunk < KCENT / CHUNK; ++chunk) {
    __syncthreads();
    // -------- Stage CHUNK centers into LDS as bf16 hi/lo + norm partials ----
    {
      const int c = tid >> 1;      // 0..127 local center
      const int h = tid & 1;       // which 32-dim half
      const float4* cr = (const float4*)(centers + (size_t)(chunk * CHUNK + c) * DIM + h * 32);
      float pn = 0.f;
#pragma unroll
      for (int j = 0; j < 8; ++j) {
        float4 t = cr[j];
        float vv[4] = {t.x, t.y, t.z, t.w};
#pragma unroll
        for (int e = 0; e < 4; ++e) {
          float vf = vv[e];
          bf16 hh = (bf16)vf;
          bf16 ll = (bf16)(vf - (float)hh);
          s_hi[c][h * 32 + j * 4 + e] = hh;
          s_lo[c][h * 32 + j * 4 + e] = ll;
          pn += vf * vf;
        }
      }
      s_pn[tid] = pn;
    }
    __syncthreads();

    // ---- 8 N-tiles of 16 centers: 1 B-fragment load feeds 12 WMMAs --------
#pragma unroll 1
    for (int t = 0; t < CHUNK / 16; ++t) {
      const int nc = t * 16 + l16;                 // this lane's center column
      const bf16* bh = &s_hi[nc][0];
      const bf16* bl = &s_lo[nc][0];
      v16bf b_hi0 = *(const v16bf*)(bh + bkoff);
      v16bf b_hi1 = *(const v16bf*)(bh + 32 + bkoff);
      v16bf b_lo0 = *(const v16bf*)(bl + bkoff);
      v16bf b_lo1 = *(const v16bf*)(bl + 32 + bkoff);

      v8f acc0 = {}, acc1 = {};
      acc0 = __builtin_amdgcn_wmma_f32_16x16x32_bf16(false, a_hi[0][0], false, b_hi0, (short)0, acc0, false, false);
      acc1 = __builtin_amdgcn_wmma_f32_16x16x32_bf16(false, a_hi[1][0], false, b_hi0, (short)0, acc1, false, false);
      acc0 = __builtin_amdgcn_wmma_f32_16x16x32_bf16(false, a_hi[0][1], false, b_hi1, (short)0, acc0, false, false);
      acc1 = __builtin_amdgcn_wmma_f32_16x16x32_bf16(false, a_hi[1][1], false, b_hi1, (short)0, acc1, false, false);
      acc0 = __builtin_amdgcn_wmma_f32_16x16x32_bf16(false, a_lo[0][0], false, b_hi0, (short)0, acc0, false, false);
      acc1 = __builtin_amdgcn_wmma_f32_16x16x32_bf16(false, a_lo[1][0], false, b_hi0, (short)0, acc1, false, false);
      acc0 = __builtin_amdgcn_wmma_f32_16x16x32_bf16(false, a_lo[0][1], false, b_hi1, (short)0, acc0, false, false);
      acc1 = __builtin_amdgcn_wmma_f32_16x16x32_bf16(false, a_lo[1][1], false, b_hi1, (short)0, acc1, false, false);
      acc0 = __builtin_amdgcn_wmma_f32_16x16x32_bf16(false, a_hi[0][0], false, b_lo0, (short)0, acc0, false, false);
      acc1 = __builtin_amdgcn_wmma_f32_16x16x32_bf16(false, a_hi[1][0], false, b_lo0, (short)0, acc1, false, false);
      acc0 = __builtin_amdgcn_wmma_f32_16x16x32_bf16(false, a_hi[0][1], false, b_lo1, (short)0, acc0, false, false);
      acc1 = __builtin_amdgcn_wmma_f32_16x16x32_bf16(false, a_hi[1][1], false, b_lo1, (short)0, acc1, false, false);

      const float cn   = s_pn[2 * nc] + s_pn[2 * nc + 1];
      const int   cidx = chunk * CHUNK + nc;
#pragma unroll
      for (int r = 0; r < 8; ++r) {
        float d0 = fabsf(xn[0][r] - 2.0f * acc0[r] + cn);
        if (d0 < best[0][r]) { best[0][r] = d0; bidx[0][r] = cidx; }  // strict < keeps first index
        float d1 = fabsf(xn[1][r] - 2.0f * acc1[r] + cn);
        if (d1 < best[1][r]) { best[1][r] = d1; bidx[1][r] = cidx; }
      }
    }
  }

  // -------- argmin reduction across the 16 lanes sharing each row ----------
#pragma unroll
  for (int mt = 0; mt < 2; ++mt) {
#pragma unroll
    for (int r = 0; r < 8; ++r) {
      float v = best[mt][r]; int i = bidx[mt][r];
#pragma unroll
      for (int off = 8; off >= 1; off >>= 1) {
        float ov = __shfl_xor(v, off, 32);
        int   oi = __shfl_xor(i, off, 32);
        if (ov < v || (ov == v && oi < i)) { v = ov; i = oi; }
      }
      best[mt][r] = v; bidx[mt][r] = i;
    }
  }

  if (l16 == 0) {
#pragma unroll
    for (int mt = 0; mt < 2; ++mt) {
      const int obase = rowbase + mt * 16 + rbase;
#pragma unroll
      for (int r = 0; r < 8; ++r) out[obase + r] = bidx[mt][r];
    }
  }
}

extern "C" void kernel_launch(void* const* d_in, const int* in_sizes, int n_in,
                              void* d_out, int out_size, void* d_ws, size_t ws_size,
                              hipStream_t stream) {
  (void)in_sizes; (void)n_in; (void)d_ws; (void)ws_size; (void)out_size;
  const float* x       = (const float*)d_in[0];
  const float* centers = (const float*)d_in[1];
  int* out             = (int*)d_out;
  dim3 grid(NPTS / ROWS_PER_BLOCK);   // 1024 blocks, 256 threads = 8 wave32
  kmeans_assign_kernel<<<grid, 256, 0, stream>>>(x, centers, out);
}
